// NTXentLoss_27848567947773
// MI455X (gfx1250) — matrix operand order repaired and checked by reading, not compiled, not run on hardware
//
#include <hip/hip_runtime.h>

// ---------------------------------------------------------------------------
// NT-Xent loss, fused: normalize -> f16 WMMA GEMM + streaming row-LSE -> mean
// gfx1250 / CDNA5: wave32, v_wmma_f32_16x16x32_f16, double-buffered B loads,
// dual accumulator chains, column-chunked grid (512 blocks).
// ---------------------------------------------------------------------------

typedef _Float16 half4v __attribute__((ext_vector_type(4)));
typedef _Float16 half8v __attribute__((ext_vector_type(8)));
typedef _Float16 v16h   __attribute__((ext_vector_type(16)));
typedef float    v8f    __attribute__((ext_vector_type(8)));

#define DIM 512
#define KT  (DIM / 32)          // 16 K-tiles of 32
#define WAVES_PER_BLOCK 8
#define ROWS_PER_BLOCK (WAVES_PER_BLOCK * 16)   // 128 rows per workgroup
#define CHUNKS 8                 // column-dimension split (grid = 64 x 8)

// ---------------------------------------------------------------------------
// Kernel 1: z = concat(z_i, z_j); z /= max(||z||, eps); store as f16.
// ---------------------------------------------------------------------------
__global__ __launch_bounds__(256)
void ntxent_normalize_f16(const float* __restrict__ zi,
                          const float* __restrict__ zj,
                          _Float16* __restrict__ z16, int Bh) {
  const int wave = (blockIdx.x * blockDim.x + threadIdx.x) >> 5;   // == row
  const int lane = threadIdx.x & 31;
  const float* src = (wave < Bh) ? (zi + (size_t)wave * DIM)
                                 : (zj + (size_t)(wave - Bh) * DIM);
  const float4* s4 = (const float4*)src;

  float4 v[4];
  float ss = 0.f;
#pragma unroll
  for (int i = 0; i < 4; ++i) {
    v[i] = s4[lane + 32 * i];
    ss += v[i].x * v[i].x + v[i].y * v[i].y + v[i].z * v[i].z + v[i].w * v[i].w;
  }
#pragma unroll
  for (int m = 16; m >= 1; m >>= 1) ss += __shfl_xor(ss, m, 32);

  const float scale = 1.0f / fmaxf(sqrtf(ss), 1e-12f);

  half4v* d4 = (half4v*)(z16 + (size_t)wave * DIM);
#pragma unroll
  for (int i = 0; i < 4; ++i) {
    half4v h;
    h[0] = (_Float16)(v[i].x * scale);
    h[1] = (_Float16)(v[i].y * scale);
    h[2] = (_Float16)(v[i].z * scale);
    h[3] = (_Float16)(v[i].w * scale);
    d4[lane + 32 * i] = h;
  }
}

// ---- helpers: load 4 B fragments / 4 WMMAs on two alternating chains -------
__device__ __forceinline__ void loadB4(v16h* dst, const _Float16* colp, int ktBase) {
#pragma unroll
  for (int i = 0; i < 4; ++i) {
    half8v b0 = *(const half8v*)(colp + (ktBase + i) * 32);
    half8v b1 = *(const half8v*)(colp + (ktBase + i) * 32 + 16);
    v16h b;
#pragma unroll
    for (int j = 0; j < 8; ++j) { b[j] = b0[j]; b[8 + j] = b1[j]; }
    dst[i] = b;
  }
}

__device__ __forceinline__ void wmma4x2(v8f& c0, v8f& c1,
                                        const v16h* Af, const v16h* Bf) {
  c0 = __builtin_amdgcn_wmma_f32_16x16x32_f16(false, Af[0], false, Bf[0],
                                              (short)0, c0, false, false);
  c1 = __builtin_amdgcn_wmma_f32_16x16x32_f16(false, Af[1], false, Bf[1],
                                              (short)0, c1, false, false);
  c0 = __builtin_amdgcn_wmma_f32_16x16x32_f16(false, Af[2], false, Bf[2],
                                              (short)0, c0, false, false);
  c1 = __builtin_amdgcn_wmma_f32_16x16x32_f16(false, Af[3], false, Bf[3],
                                              (short)0, c1, false, false);
}

// ---------------------------------------------------------------------------
// Kernel 2: fused sim-GEMM + streaming Sigma-exp per row, one column chunk.
// ---------------------------------------------------------------------------
__global__ __launch_bounds__(256)
void ntxent_lse_wmma(const _Float16* __restrict__ z16,
                     float* __restrict__ sums, float* __restrict__ posArr,
                     int Bh, int N) {
  const float invT = 10.0f;        // 1 / 0.1
  const int lane = threadIdx.x & 31;
  const int wave = threadIdx.x >> 5;
  const int rowBase = (blockIdx.x * WAVES_PER_BLOCK + wave) * 16;
  const int tilesPerChunk = (N / 16) / CHUNKS;
  const int chunkStart = blockIdx.y * tilesPerChunk * 16;

  // ---- preload A fragments (16x32 f16 per K-tile, all K resident) ---------
  v16h A[KT];
  {
    const int m  = lane & 15;
    const int hk = (lane >> 4) * 8;
    const _Float16* rowp = z16 + (size_t)(rowBase + m) * DIM;
#pragma unroll
    for (int kt = 0; kt < KT; ++kt) {
      half8v c0 = *(const half8v*)(rowp + kt * 32 + hk);
      half8v c1 = *(const half8v*)(rowp + kt * 32 + hk + 16);
      v16h a;
#pragma unroll
      for (int i = 0; i < 8; ++i) { a[i] = c0[i]; a[8 + i] = c1[i]; }
      A[kt] = a;
    }
  }

  float acc[8];
#pragma unroll
  for (int r = 0; r < 8; ++r) acc[r] = 0.f;
  float pv = 0.f;
  const int posColBase = (rowBase < Bh) ? (rowBase + Bh) : (rowBase - Bh);
  const bool ownsPos = (posColBase >= chunkStart) &&
                       (posColBase <  chunkStart + tilesPerChunk * 16);

  const int n      = lane & 15;          // B/C column owned by this lane
  const int kgrp   = (lane >> 4) * 8;    // B K-chunk per lane group
  const int rowOff = (lane >> 4) * 8;    // C rows: VGPR r -> M = r + rowOff

  // ---- double-buffered column sweep ---------------------------------------
  v16h B0[4], B1[4];
  const _Float16* colp = z16 + (size_t)(chunkStart + n) * DIM + kgrp;
  loadB4(B0, colp, 0);                   // prologue: kt 0..3 of first tile
  int colBase = chunkStart;

  for (int t = 0; t < tilesPerChunk; ++t) {
    const int nextCol = (t + 1 < tilesPerChunk) ? (colBase + 16) : chunkStart;
    const _Float16* colpN = z16 + (size_t)(nextCol + n) * DIM + kgrp;

    v8f c0 = {}, c1 = {};
    loadB4(B1, colp, 4);
    wmma4x2(c0, c1, &A[0],  B0);
    loadB4(B0, colp, 8);
    wmma4x2(c0, c1, &A[4],  B1);
    loadB4(B1, colp, 12);
    wmma4x2(c0, c1, &A[8],  B0);
    loadB4(B0, colpN, 0);                // prefetch next tile's kt 0..3
    wmma4x2(c0, c1, &A[12], B1);

    v8f c;
#pragma unroll
    for (int r = 0; r < 8; ++r) c[r] = c0[r] + c1[r];

    // epilogue: accumulate exp; diagonal exists only in the colBase==rowBase
    // tile (both 16-aligned), so mask work is hoisted to a uniform branch.
    if (colBase == rowBase) {
      const int gcol = colBase + n;
#pragma unroll
      for (int r = 0; r < 8; ++r) {
        const int grow = rowBase + rowOff + r;
        const float e = __expf(c[r] * invT);
        acc[r] += (grow == gcol) ? 0.f : e;
      }
    } else {
#pragma unroll
      for (int r = 0; r < 8; ++r) acc[r] += __expf(c[r] * invT);
    }

    // capture positives (tile diagonal of the rowBase +/- B column tile)
    if (colBase == posColBase) {
#pragma unroll
      for (int r = 0; r < 8; ++r) {
        if (lane == r)      pv = c[r] * invT;   // rows 0..7
        if (lane == 24 + r) pv = c[r] * invT;   // rows 8..15
      }
    }
    colBase = nextCol;
    colp = colpN;
  }

  // ---- reduce Sigma-exp across the 16 lanes sharing each row --------------
#pragma unroll
  for (int r = 0; r < 8; ++r) {
#pragma unroll
    for (int m = 8; m >= 1; m >>= 1) acc[r] += __shfl_xor(acc[r], m, 32);
  }

  float* sumsC = sums + (size_t)blockIdx.y * N;
  if (lane == 0) {
#pragma unroll
    for (int r = 0; r < 8; ++r) sumsC[rowBase + r] = acc[r];
  }
  if (lane == 16) {
#pragma unroll
    for (int r = 0; r < 8; ++r) sumsC[rowBase + 8 + r] = acc[r];
  }
  // dense positive store: exactly one chunk owns each row's positive
  if (ownsPos) {
    if (lane < 8)   posArr[rowBase + lane] = pv;
    if (lane >= 24) posArr[rowBase + 8 + (lane - 24)] = pv;
  }
}

// ---------------------------------------------------------------------------
// Kernel 3: per-row combine across chunks, per-block deterministic reduce.
// ---------------------------------------------------------------------------
__global__ __launch_bounds__(128)
void ntxent_rowloss(const float* __restrict__ sums, const float* __restrict__ posArr,
                    float* __restrict__ blkpart, int N) {
  __shared__ float sRed[128];
  const int tid = threadIdx.x;
  const int row = blockIdx.x * 128 + tid;
  float s = 0.f;
#pragma unroll
  for (int c = 0; c < CHUNKS; ++c) s += sums[(size_t)c * N + row];
  sRed[tid] = __logf(s) - posArr[row];
  __syncthreads();
#pragma unroll
  for (int k = 64; k > 0; k >>= 1) {
    if (tid < k) sRed[tid] += sRed[tid + k];
    __syncthreads();
  }
  if (tid == 0) blkpart[blockIdx.x] = sRed[0];
}

// ---------------------------------------------------------------------------
// Kernel 4: fixed-order final mean (no atomics -> deterministic).
// ---------------------------------------------------------------------------
__global__ void ntxent_finish(const float* __restrict__ partial,
                              float* __restrict__ out, int nb, int N) {
  __shared__ float s[64];
  const int t = threadIdx.x;
  s[t] = (t < nb) ? partial[t] : 0.f;
  __syncthreads();
#pragma unroll
  for (int k = 32; k > 0; k >>= 1) {
    if (t < k) s[t] += s[t + k];
    __syncthreads();
  }
  if (t == 0) out[0] = s[0] / (float)N;
}

// ---------------------------------------------------------------------------
extern "C" void kernel_launch(void* const* d_in, const int* in_sizes, int n_in,
                              void* d_out, int out_size, void* d_ws, size_t ws_size,
                              hipStream_t stream) {
  (void)n_in; (void)out_size; (void)ws_size;
  const float* zi = (const float*)d_in[0];
  const float* zj = (const float*)d_in[1];
  const int Bh = in_sizes[0] / DIM;   // 4096
  const int N  = 2 * Bh;              // 8192

  char* ws = (char*)d_ws;
  _Float16* z16   = (_Float16*)ws;    ws += (size_t)N * DIM * sizeof(_Float16);
  float* sums     = (float*)ws;       ws += (size_t)CHUNKS * N * sizeof(float);
  float* posArr   = (float*)ws;       ws += (size_t)N * sizeof(float);
  float* blkpart  = (float*)ws;
  float* out      = (float*)d_out;

  // 1) normalize + pack f16 (one wave per row, 8 waves per block)
  ntxent_normalize_f16<<<N / 8, 256, 0, stream>>>(zi, zj, z16, Bh);

  // 2) fused WMMA GEMM + row Sigma-exp: 64 row-blocks x 8 column chunks
  dim3 g2(N / ROWS_PER_BLOCK, CHUNKS);
  ntxent_lse_wmma<<<g2, 256, 0, stream>>>(z16, sums, posArr, Bh, N);

  // 3) per-row loss combine + block partials (64 blocks)
  ntxent_rowloss<<<N / 128, 128, 0, stream>>>(sums, posArr, blkpart, N);

  // 4) deterministic final mean
  ntxent_finish<<<1, 64, 0, stream>>>(blkpart, out, N / 128, N);
}
